// HGCNMixer_31885837205682
// MI455X (gfx1250) — matrix operand order repaired and checked by reading, not compiled, not run on hardware
//
#include <hip/hip_runtime.h>
#include <hip/hip_bf16.h>

// ---------------------------------------------------------------------------
// HGCN mixer for MI455X (gfx1250).  bf16 WMMA (v_wmma_f32_16x16x32_bf16) for
// all GEMMs, TDM (tensor_load_to_lds) double-buffered staging for the big
// streaming GEMM input, fp32 VALU for the tiny HGCN mat-vec chains.
// ---------------------------------------------------------------------------

typedef __attribute__((ext_vector_type(16))) __bf16 v16bf;
typedef __attribute__((ext_vector_type(8)))  float  v8f;

constexpr int NB  = 32 * 512;  // flattened batch
constexpr int NA  = 32;        // agents
constexpr int OBS = 96;        // obs dim
constexpr int SD  = 1024;      // state dim
constexpr int NE  = 64;        // edges
constexpr int HD  = 256;       // hidden per MLP

#if defined(__has_builtin)
#  if __has_builtin(__builtin_amdgcn_tensor_load_to_lds) && \
      __has_builtin(__builtin_amdgcn_s_wait_tensorcnt)
#    define HAVE_TDM 1
#  else
#    define HAVE_TDM 0
#  endif
#else
#  define HAVE_TDM 0
#endif

static __device__ inline unsigned short f2bf(float f) {
    union { float f; unsigned int u; } v; v.f = f;
    unsigned int u = v.u;
    unsigned int r = u + 0x7FFFu + ((u >> 16) & 1u);   // round-to-nearest-even
    return (unsigned short)(r >> 16);
}

static __device__ inline v8f vzero8() {
    v8f z = {0.f, 0.f, 0.f, 0.f, 0.f, 0.f, 0.f, 0.f};
    return z;
}

union FragU { v16bf v; unsigned int u[8]; };

// A fragment (16x32 bf16) from row-major bf16 storage.
static __device__ inline v16bf load_a_frag(const unsigned short* base, int stride,
                                           int mbase, int koff, int lane) {
    FragU f;
    const int m = mbase + (lane & 15);
    const int half = lane >> 4;
    const unsigned int* row =
        reinterpret_cast<const unsigned int*>(base + (size_t)m * stride);
#pragma unroll
    for (int j = 0; j < 8; ++j) {
        const int k0 = koff + ((j >> 2) << 4) + (half << 3) + ((j & 3) << 1);
        f.u[j] = row[k0 >> 1];
    }
    return f.v;
}

// A fragment (16x32 bf16) built on the fly from row-major fp32 LDS (TDM tile).
static __device__ inline v16bf load_a_frag_f32(const float* base, int stride,
                                               int mbase, int koff, int lane) {
    FragU f;
    const int m = mbase + (lane & 15);
    const int half = lane >> 4;
    const float* row = base + (size_t)m * stride;
#pragma unroll
    for (int j = 0; j < 8; ++j) {
        const int k0 = koff + ((j >> 2) << 4) + (half << 3) + ((j & 3) << 1);
        const unsigned lo = f2bf(row[k0]);
        const unsigned hi = f2bf(row[k0 + 1]);
        f.u[j] = lo | (hi << 16);
    }
    return f.v;
}

// B fragment (32x16 bf16) from COLUMN-major bf16 storage (pre-swizzled).
static __device__ inline v16bf load_b_frag(const unsigned short* base, int stride,
                                           int nbase, int koff, int lane) {
    FragU f;
    const int n = nbase + (lane & 15);
    const int half = lane >> 4;
    const unsigned int* col =
        reinterpret_cast<const unsigned int*>(base + (size_t)n * stride);
#pragma unroll
    for (int j = 0; j < 8; ++j) {
        const int k = koff + ((j >> 2) << 4) + (half << 3) + ((j & 3) << 1);
        f.u[j] = col[k >> 1];
    }
    return f.v;
}

static __device__ inline v8f wmma_bf16(v16bf a, v16bf b, v8f c) {
    return __builtin_amdgcn_wmma_f32_16x16x32_bf16(
        /*neg_a=*/false, a, /*neg_b=*/false, b,
        /*c_mod=*/(short)0, c, /*reuse_a=*/false, /*reuse_b=*/false);
}

#if HAVE_TDM
// ---------------------------------------------------------------------------
// TDM: 2D fp32 tile (tile_w x tile_h elements) from global to LDS.
// D# per CDNA5 ISA ch.8: group0 {count=1, lds_addr, global_addr, type=2},
// group1 {data_size=4B, pad 1 dword per 32 dwords, tensor=tile dims, stride}.
// Padding makes LDS rows land at a 33-dword stride (bank-conflict free).
// This toolchain's builtin takes 6 args (g0,g1,g2,g3,g4,cpol).
// ---------------------------------------------------------------------------
static __device__ inline void tdm_load_2d_f32(unsigned lds_byte_addr,
                                              const float* gptr,
                                              int tile_w, int tile_h,
                                              int row_stride_elems) {
    typedef __attribute__((ext_vector_type(4))) unsigned int v4u;
    typedef __attribute__((ext_vector_type(8))) int v8i;
    typedef __attribute__((ext_vector_type(4))) int v4i;
    const unsigned long long ga = (unsigned long long)(size_t)gptr;
    v4u g0;
    g0[0] = 1u;                                            // count=1, user mode
    g0[1] = lds_byte_addr;                                 // lds_addr
    g0[2] = (unsigned)(ga & 0xFFFFFFFFull);                // global_addr[31:0]
    g0[3] = (unsigned)((ga >> 32) & 0x01FFFFFFull) | (0x2u << 30);  // type=2
    v8i g1;
    g1[0] = (int)((2u << 16)      // data_size = 4B
                | (1u << 20)      // pad_enable
                | (4u << 22)      // pad_interval code 4 = 32 DWORDs
                | (0u << 25));    // pad_amount  code 0 = 1 DWORD
    g1[1] = (int)(((unsigned)tile_w & 0xFFFFu) << 16);     // tensor_dim0[15:0]
    g1[2] = (int)(((unsigned)tile_h & 0xFFFFu) << 16);     // tensor_dim1[15:0]
    g1[3] = (int)(((unsigned)tile_w & 0xFFFFu) << 16);     // tile_dim0
    g1[4] = (int)((unsigned)tile_h & 0xFFFFu);             // tile_dim1, tile_dim2=0
    g1[5] = (int)(unsigned)row_stride_elems;               // tensor_dim0_stride
    g1[6] = 0;
    g1[7] = 0;
    const v4i gz4 = {0, 0, 0, 0};
    const v8i gz8 = {0, 0, 0, 0, 0, 0, 0, 0};
    __builtin_amdgcn_tensor_load_to_lds(g0, g1, gz4, gz4, gz8, 0);
}
#endif

// ---------------------------------------------------------------------------
// Kernel 1: hidden = relu(states @ W1cat + b1cat)   [B,1024] bf16 out
// 64x64 tile per 256-thread block; each of 8 waves owns a 16x64 strip.
// S tile staged by TDM (fp32, double buffered), W tile staged manually (bf16,
// transposed, double buffered).
// ---------------------------------------------------------------------------
__global__ __launch_bounds__(256) void k_hidden(
    const float* __restrict__ S,
    const float* __restrict__ Wa, const float* __restrict__ Ba,
    const float* __restrict__ Wb, const float* __restrict__ Bb,
    const float* __restrict__ Wc, const float* __restrict__ Bc,
    const float* __restrict__ Wd, const float* __restrict__ Bd,
    unsigned short* __restrict__ hidden)
{
    constexpr int LSS = 33;   // fp32 S row stride in LDS (32 + 1 TDM pad dword)
    constexpr int LDB = 36;   // bf16 W column stride in LDS
    constexpr int KT  = SD / 32;
    __shared__ float          sS[2][64 * LSS];
    __shared__ unsigned short sBt[2][64 * LDB];

    const int tid = threadIdx.x;
    const int wv = tid >> 5, lane = tid & 31;
    const int row0 = blockIdx.x * 64;
    const int mlp = blockIdx.y >> 2;          // which of the 4 MLPs
    const int col0 = (blockIdx.y & 3) * 64;   // column offset within its HID=256
    const float* Wp = (mlp == 0) ? Wa : (mlp == 1) ? Wb : (mlp == 2) ? Wc : Wd;
    const float* Bp = (mlp == 0) ? Ba : (mlp == 1) ? Bb : (mlp == 2) ? Bc : Bd;
    const float* Sblk = S + (size_t)row0 * SD;

    v8f acc[4];
#pragma unroll
    for (int i = 0; i < 4; ++i) acc[i] = vzero8();

    // ---- stage tile 0 ----
#if HAVE_TDM
    if (wv == 0)
        tdm_load_2d_f32((unsigned)(size_t)&sS[0][0], Sblk, 32, 64, SD);
#else
    for (int i = tid; i < 64 * 32; i += 256) {
        const int r = i >> 5, c = i & 31;
        sS[0][r * LSS + c] = Sblk[(size_t)r * SD + c];
    }
#endif
    for (int i = tid; i < 32 * 64; i += 256) {
        const int k = i >> 6, n = i & 63;
        sBt[0][n * LDB + k] = f2bf(Wp[(size_t)k * HD + col0 + n]);
    }
#if HAVE_TDM
    if (wv == 0) __builtin_amdgcn_s_wait_tensorcnt(0);
#endif
    __syncthreads();

    for (int kt = 0; kt < KT; ++kt) {
        const int cur = kt & 1;
        if (kt + 1 < KT) {
            const int k1 = (kt + 1) * 32;
#if HAVE_TDM
            if (wv == 0)
                tdm_load_2d_f32((unsigned)(size_t)&sS[cur ^ 1][0],
                                Sblk + k1, 32, 64, SD);
#else
            for (int i = tid; i < 64 * 32; i += 256) {
                const int r = i >> 5, c = i & 31;
                sS[cur ^ 1][r * LSS + c] = Sblk[(size_t)r * SD + k1 + c];
            }
#endif
            for (int i = tid; i < 32 * 64; i += 256) {
                const int k = i >> 6, n = i & 63;
                sBt[cur ^ 1][n * LDB + k] = f2bf(Wp[(size_t)(k1 + k) * HD + col0 + n]);
            }
            if (kt + 2 < KT)   // gfx1250 global_prefetch_b8 for the W slab
                __builtin_prefetch(&Wp[(size_t)(k1 + 32) * HD + col0 + (tid & 63)], 0, 1);
        }

        const v16bf a = load_a_frag_f32(sS[cur], LSS, wv * 16, 0, lane);
#pragma unroll
        for (int nt = 0; nt < 4; ++nt) {
            const v16bf b = load_b_frag(sBt[cur], LDB, nt * 16, 0, lane);
            acc[nt] = wmma_bf16(a, b, acc[nt]);
        }
#if HAVE_TDM
        if (wv == 0 && kt + 1 < KT) __builtin_amdgcn_s_wait_tensorcnt(0);
#endif
        __syncthreads();
    }

    const int half = lane >> 4, nl = lane & 15;
#pragma unroll
    for (int nt = 0; nt < 4; ++nt) {
        const int cl = col0 + nt * 16 + nl;   // col within this MLP's 256
        const float bias = Bp[cl];
        const int gn = mlp * HD + cl;         // col within fused 1024
#pragma unroll
        for (int i = 0; i < 8; ++i) {
            const int gr = row0 + wv * 16 + i + 8 * half;
            float v = acc[nt][i] + bias;
            v = v > 0.f ? v : 0.f;
            hidden[(size_t)gr * (4 * HD) + gn] = f2bf(v);
        }
    }
}

// ---------------------------------------------------------------------------
// Kernel 2: head MLPs second layer.  w1=|hw1|, c1=hc1, w=|hw|, c=hc.
// 256 threads = 8 waves; each wave handles 16 batch rows.
// ---------------------------------------------------------------------------
__global__ __launch_bounds__(256) void k_heads(
    const unsigned short* __restrict__ hidden,
    const float* __restrict__ w2a, const float* __restrict__ b2a,   // hw1
    const float* __restrict__ w2b, const float* __restrict__ b2b,   // hc1
    const float* __restrict__ w2c, const float* __restrict__ b2c,   // hw
    const float* __restrict__ w2d, const float* __restrict__ b2d,   // hc
    float* __restrict__ w1a, float* __restrict__ c1v,
    float* __restrict__ wav, float* __restrict__ ccv)
{
    constexpr int LDW = 258;                       // 256 + pad, even
    constexpr int HCOFF = 3 * 32 * LDW;
    __shared__ unsigned short sW[3 * 32 * LDW + 16 * LDW];

    const int tid = threadIdx.x;
    // stage three 256x32 W2 matrices, bf16, column-major
    for (int i = tid; i < 3 * HD * 32; i += 256) {
        const int m = i / (HD * 32);
        const int rem = i - m * (HD * 32);
        const int k = rem >> 5, n = rem & 31;
        const float* wp = (m == 0) ? w2a : (m == 1) ? w2b : w2c;
        sW[m * 32 * LDW + n * LDW + k] = f2bf(wp[k * 32 + n]);
    }
    // hc weight: col 0 = w2d, cols 1..15 zero (N padded to 16)
    for (int i = tid; i < 15 * LDW; i += 256) {
        const int n = 1 + i / LDW;
        const int k = i - (n - 1) * LDW;
        sW[HCOFF + n * LDW + k] = 0;
    }
    for (int i = tid; i < LDW; i += 256)
        sW[HCOFF + i] = (i < HD) ? f2bf(w2d[i]) : (unsigned short)0;
    __syncthreads();

    const int wv = tid >> 5, lane = tid & 31;
    const int r0 = blockIdx.x * 128 + wv * 16;
    const unsigned short* arow = hidden + (size_t)r0 * (4 * HD);
    const int half = lane >> 4, nl = lane & 15;

#pragma unroll
    for (int m = 0; m < 3; ++m) {
        v8f a0 = vzero8(), a1 = vzero8();
        for (int kt = 0; kt < 8; ++kt) {
            const v16bf a = load_a_frag(arow, 4 * HD, 0, m * HD + kt * 32, lane);
            const v16bf b0 = load_b_frag(sW + m * 32 * LDW, LDW, 0, kt * 32, lane);
            const v16bf b1 = load_b_frag(sW + m * 32 * LDW, LDW, 16, kt * 32, lane);
            a0 = wmma_bf16(a, b0, a0);
            a1 = wmma_bf16(a, b1, a1);
        }
        const float* bp = (m == 0) ? b2a : (m == 1) ? b2b : b2c;
        float* op = (m == 0) ? w1a : (m == 1) ? c1v : wav;
        const bool doabs = (m != 1);
        const float bi0 = bp[nl], bi1 = bp[16 + nl];
#pragma unroll
        for (int i = 0; i < 8; ++i) {
            const int r = r0 + i + 8 * half;
            float v0 = a0[i] + bi0, v1 = a1[i] + bi1;
            if (doabs) { v0 = fabsf(v0); v1 = fabsf(v1); }
            op[(size_t)r * NA + nl] = v0;
            op[(size_t)r * NA + 16 + nl] = v1;
        }
    }
    // hc head (scalar output, N padded to 16)
    v8f ac = vzero8();
    for (int kt = 0; kt < 8; ++kt) {
        const v16bf a = load_a_frag(arow, 4 * HD, 0, 3 * HD + kt * 32, lane);
        const v16bf b = load_b_frag(sW + HCOFF, LDW, 0, kt * 32, lane);
        ac = wmma_bf16(a, b, ac);
    }
    if (nl == 0) {
        const float bi = b2d[0];
#pragma unroll
        for (int i = 0; i < 8; ++i)
            ccv[r0 + i + 8 * half] = ac[i] + bi;
    }
}

// ---------------------------------------------------------------------------
// Kernel 3: per batch row: H = relu(u @ edge_W + b) via WMMA, then two HGCN
// levels (fp32 VALU), elu mix, wave32 reduction.  64 threads = 2 waves,
// one batch row per wave.
// ---------------------------------------------------------------------------
__global__ __launch_bounds__(64) void k_mix(
    const float* __restrict__ agent_qs, const float* __restrict__ U,
    const float* __restrict__ edge_W, const float* __restrict__ edge_b,
    const float* __restrict__ wline1, const float* __restrict__ wline2,
    const float* __restrict__ w1a, const float* __restrict__ c1v,
    const float* __restrict__ wav, const float* __restrict__ ccv,
    float* __restrict__ out)
{
    constexpr int LEW = 98, LU = 98, LH = 66;
    __shared__ unsigned short sEW[NE * LEW];      // edge_W, bf16, column-major
    __shared__ unsigned short sU[2][NA * LU];     // u tile per wave, bf16
    __shared__ float sH[2][NA * LH];              // H per wave, fp32
    __shared__ float sWA[2][NE];                  // |wline1|, |wline2|
    __shared__ float sT[2][NA];
    __shared__ float sS2[2][NE];

    const int tid = threadIdx.x;
    const int wv = tid >> 5, lane = tid & 31;
    const size_t b = (size_t)blockIdx.x * 2 + wv;

    for (int i = tid; i < OBS * NE; i += 64) {
        const int k = i >> 6, n = i & 63;
        sEW[n * LEW + k] = f2bf(edge_W[k * NE + n]);
    }
    sWA[0][tid] = fabsf(wline1[tid]);   // blockDim 64 == NE
    sWA[1][tid] = fabsf(wline2[tid]);
    for (int i = lane; i < NA * OBS; i += 32) {
        const int n = i / OBS, k = i - n * OBS;
        sU[wv][n * LU + k] = f2bf(U[b * (NA * OBS) + i]);
    }
    __syncthreads();

    const int half = lane >> 4, nl = lane & 15;
#pragma unroll
    for (int mt = 0; mt < 2; ++mt) {
        v8f acc[4];
#pragma unroll
        for (int i = 0; i < 4; ++i) acc[i] = vzero8();
#pragma unroll
        for (int kt = 0; kt < 3; ++kt) {          // K = 96 = 3 * 32
            const v16bf a = load_a_frag(sU[wv], LU, mt * 16, kt * 32, lane);
#pragma unroll
            for (int nt = 0; nt < 4; ++nt) {
                const v16bf bb = load_b_frag(sEW, LEW, nt * 16, kt * 32, lane);
                acc[nt] = wmma_bf16(a, bb, acc[nt]);
            }
        }
#pragma unroll
        for (int nt = 0; nt < 4; ++nt) {
            const int n = nt * 16 + nl;
            const float eb = edge_b[n];
#pragma unroll
            for (int i = 0; i < 8; ++i) {
                const int mm = mt * 16 + i + 8 * half;
                const float v = acc[nt][i] + eb;
                sH[wv][mm * LH + n] = v > 0.f ? v : 0.f;
            }
        }
    }
    __syncthreads();

    const float* Hrow = &sH[wv][lane * LH];       // lane <-> agent
    // b_e (shared by both levels); this lane owns edges e=lane, e=lane+32
    float be0 = 0.f, be1 = 0.f;
    for (int n = 0; n < NA; ++n) {
        be0 += sH[wv][n * LH + lane];
        be1 += sH[wv][n * LH + lane + 32];
    }
    const float binv0 = be0 > 0.f ? 1.f / be0 : 0.f;
    const float binv1 = be1 > 0.f ? 1.f / be1 : 0.f;

    float x = agent_qs[b * NA + lane];
#pragma unroll
    for (int lvl = 0; lvl < 2; ++lvl) {
        float d = 0.f;
        for (int e = 0; e < NE; ++e) d += Hrow[e] * sWA[lvl][e];
        const float dis = d > 0.f ? rsqrtf(d) : 0.f;
        sT[wv][lane] = dis * x;
        __syncthreads();
        float s0 = 0.f, s1 = 0.f;
        for (int n = 0; n < NA; ++n) {
            const float tn = sT[wv][n];
            s0 += sH[wv][n * LH + lane] * tn;
            s1 += sH[wv][n * LH + lane + 32] * tn;
        }
        sS2[wv][lane]      = s0 * sWA[lvl][lane] * binv0;
        sS2[wv][lane + 32] = s1 * sWA[lvl][lane + 32] * binv1;
        __syncthreads();
        float y = 0.f;
        for (int e = 0; e < NE; ++e) y += Hrow[e] * sS2[wv][e];
        x = dis * y;
        __syncthreads();
    }

    const float w1 = w1a[b * NA + lane];
    const float c1 = c1v[b * NA + lane];
    const float w  = wav[b * NA + lane];
    const float z  = x * w1 + c1;
    const float qt = z > 0.f ? z : (expf(z) - 1.f);  // elu
    float contrib = qt * w;
#pragma unroll
    for (int off = 16; off > 0; off >>= 1)
        contrib += __shfl_xor(contrib, off, 32);
    if (lane == 0) out[b] = contrib + ccv[b];
}

// ---------------------------------------------------------------------------
extern "C" void kernel_launch(void* const* d_in, const int* in_sizes, int n_in,
                              void* d_out, int out_size, void* d_ws, size_t ws_size,
                              hipStream_t stream)
{
    const float* agent_qs = (const float*)d_in[0];
    const float* states   = (const float*)d_in[1];
    const float* indiv_us = (const float*)d_in[2];
    const float* edge_W   = (const float*)d_in[3];
    const float* edge_b   = (const float*)d_in[4];
    const float* wline1   = (const float*)d_in[5];
    const float* wline2   = (const float*)d_in[6];
    const float* hw1_w1 = (const float*)d_in[7];
    const float* hw1_b1 = (const float*)d_in[8];
    const float* hw1_w2 = (const float*)d_in[9];
    const float* hw1_b2 = (const float*)d_in[10];
    const float* hc1_w1 = (const float*)d_in[11];
    const float* hc1_b1 = (const float*)d_in[12];
    const float* hc1_w2 = (const float*)d_in[13];
    const float* hc1_b2 = (const float*)d_in[14];
    const float* hw_w1  = (const float*)d_in[15];
    const float* hw_b1  = (const float*)d_in[16];
    const float* hw_w2  = (const float*)d_in[17];
    const float* hw_b2  = (const float*)d_in[18];
    const float* hc_w1  = (const float*)d_in[19];
    const float* hc_b1  = (const float*)d_in[20];
    const float* hc_w2  = (const float*)d_in[21];
    const float* hc_b2  = (const float*)d_in[22];
    float* out = (float*)d_out;

    // workspace: hidden bf16 [NB,1024], then fp32 head outputs
    unsigned short* hidden = (unsigned short*)d_ws;
    char* base = (char*)d_ws;
    float* w1a = (float*)(base + (size_t)NB * (4 * HD) * sizeof(unsigned short));
    float* c1v = w1a + (size_t)NB * NA;
    float* wav = c1v + (size_t)NB * NA;
    float* ccv = wav + (size_t)NB * NA;

    k_hidden<<<dim3(NB / 64, 16), 256, 0, stream>>>(
        states, hw1_w1, hw1_b1, hc1_w1, hc1_b1, hw_w1, hw_b1, hc_w1, hc_b1, hidden);

    k_heads<<<dim3(NB / 128), 256, 0, stream>>>(
        hidden, hw1_w2, hw1_b2, hc1_w2, hc1_b2, hw_w2, hw_b2, hc_w2, hc_b2,
        w1a, c1v, wav, ccv);

    k_mix<<<dim3(NB / 2), 64, 0, stream>>>(
        agent_qs, indiv_us, edge_W, edge_b, wline1, wline2,
        w1a, c1v, wav, ccv, out);

    (void)in_sizes; (void)n_in; (void)out_size; (void)ws_size;
}
